// DocumentLevelAttention_24885040513253
// MI455X (gfx1250) — matrix operand order repaired and checked by reading, not compiled, not run on hardware
//
#include <hip/hip_runtime.h>
#include <hip/hip_bf16.h>
#include <math.h>
#include <stdint.h>

// ---------------------------------------------------------------------------
// Problem constants (reference: B=16, L=1024, d_model=1024, 4 heads, d_k=256)
// ---------------------------------------------------------------------------
#define D_MODEL 1024
#define SEQ_L   1024
#define BATCH   16
#define N_HEAD  4
#define D_K     256
#define MROWS   (BATCH * SEQ_L)             // 16384
#define OUT1_ELEMS ((long)MROWS * D_MODEL)  // 16,777,216 floats

// ---- CDNA5 async / TDM feature detection (device pass only) ----
#if defined(__HIP_DEVICE_COMPILE__)
#if __has_builtin(__builtin_amdgcn_tensor_load_to_lds) && \
    __has_builtin(__builtin_amdgcn_s_wait_tensorcnt)
#define HAVE_TDM 1
#endif
#if __has_builtin(__builtin_amdgcn_global_load_async_to_lds_b128) && \
    __has_builtin(__builtin_amdgcn_s_wait_asynccnt)
#define HAVE_ASYNC 1
#endif
#endif

typedef __attribute__((ext_vector_type(16))) _Float16 v16h;
typedef __attribute__((ext_vector_type(8)))  _Float16 v8h;
typedef __attribute__((ext_vector_type(8)))  float    v8f;

union Frag { v16h v; v8h h[2]; };

__device__ __forceinline__ v8f wmma16(v16h a, v16h b, v8f c) {
    return __builtin_amdgcn_wmma_f32_16x16x32_f16(
        false, a, false, b, (short)0, c, false, false);
}

__device__ __forceinline__ v8f zero8() {
    v8f z = {0.f,0.f,0.f,0.f,0.f,0.f,0.f,0.f};
    return z;
}

#ifdef HAVE_ASYNC
#define AS1 __attribute__((address_space(1)))
#define AS3 __attribute__((address_space(3)))
typedef int v4i __attribute__((ext_vector_type(4)));
// toolchain signature: (int4 AS1* gsrc, int4 AS3* ldst, imm offset, imm cpol)
__device__ __forceinline__ void async_b128(const void* g, void* l) {
    __builtin_amdgcn_global_load_async_to_lds_b128(
        (AS1 v4i*)g, (AS3 v4i*)l, 0, 0);
}
#endif

#ifdef HAVE_TDM
typedef unsigned int u32x4 __attribute__((ext_vector_type(4)));
typedef int          i32x4 __attribute__((ext_vector_type(4)));
typedef int          i32x8 __attribute__((ext_vector_type(8)));

// 2-D f16 tile DMA: tile_d1 rows of tile_d0 elements, row stride = stride_d0
// elements. Descriptor packed per CDNA5 ISA 8.3/8.4 (count=1, data_size=2B,
// type=2 "image"); groups 2/3 zero (<=2-D tensor). 6-arg clang-23 builtin.
__device__ __forceinline__ void tdm_load_2d_f16(const void* gsrc, unsigned lds_off,
                                                unsigned tile_d0, unsigned tile_d1,
                                                unsigned stride_d0) {
    unsigned long long ga = (unsigned long long)(uintptr_t)gsrc;
    const unsigned td0 = stride_d0;     // tensor_dim0 (elements)
    const unsigned td1 = 1u << 20;      // tensor_dim1: large, never clips
    u32x4 g0;
    g0[0] = 1u;                                           // count = 1 descriptor
    g0[1] = lds_off;                                      // lds_addr
    g0[2] = (unsigned)ga;                                 // global_addr[31:0]
    g0[3] = (unsigned)((ga >> 32) & 0x1FFFFFFu) | 0x80000000u; // [56:32]|type=2
    i32x8 g1;
    g1[0] = (int)(1u << 16);                              // data_size = 2 bytes
    g1[1] = (int)((td0 & 0xFFFFu) << 16);                 // tensor_dim0[15:0]
    g1[2] = (int)((td0 >> 16) | ((td1 & 0xFFFFu) << 16)); // td0 hi | td1 lo
    g1[3] = (int)((td1 >> 16) | (tile_d0 << 16));         // td1 hi | tile_dim0
    g1[4] = (int)(tile_d1 & 0xFFFFu);                     // tile_dim1 (tile_dim2=0)
    g1[5] = (int)stride_d0;                               // tensor_dim0_stride lo
    g1[6] = 0;                                            // stride hi | dim1_stride lo
    g1[7] = 0;
    i32x4 z4 = {0,0,0,0};
    i32x8 z8 = {0,0,0,0,0,0,0,0};
    __builtin_amdgcn_tensor_load_to_lds(g0, g1, z4, z4, z8, 0);
}
#endif

// ---------------------------------------------------------------------------
// Weight transpose + f32 -> f16 : src[h][r][c] f32  ->  dst[h][c][r] f16
// ---------------------------------------------------------------------------
__global__ __launch_bounds__(256) void k_transpose_cvt(
    const float* __restrict__ src, _Float16* __restrict__ dst,
    int H, int R, int C) {
    long i = (long)blockIdx.x * blockDim.x + threadIdx.x;
    long total = (long)H * R * C;
    if (i >= total) return;
    int c = (int)(i % C);
    long t = i / C;
    int r = (int)(t % R);
    int h = (int)(t / R);
    dst[((long)h * C + c) * R + r] = (_Float16)src[i];
}

// ---------------------------------------------------------------------------
// WMMA GEMM:  C[M x N] = A[M x 1024] * BT[N x 1024]^T
// BM=128, BN=64, BK=32, 256 threads = 8 waves.
// With TDM: the full 64x1024 f16 B panel (128KB of the 320KB WGP LDS) is
// DMA'd once per block by a single TENSOR_LOAD_TO_LDS; the K loop then only
// stages A. Without TDM: per-step B staging.
// MODE 0: f16 q_s/k_s  [h][m][n&255]   MODE 1: f16 v_t [h][b][n&255][lk]
// MODE 2: f32 Z = acc + bias[n] + resid[m][n]
// ---------------------------------------------------------------------------
template<int MODE, bool AF32>
__global__ __launch_bounds__(256) void k_gemm(
    const void* __restrict__ Aptr, const _Float16* __restrict__ BT,
    float* __restrict__ Cf, _Float16* __restrict__ Ch,
    const float* __restrict__ bias, const float* __restrict__ resid) {

    constexpr int K = 1024;
    __shared__ _Float16 Al[128 * 32];
    extern __shared__ _Float16 Bp[];   // TDM: 64x1024 panel; else 64x32 buffer

    const int tid  = threadIdx.x;
    const int w    = tid >> 5;
    const int lane = tid & 31;
    const int lm   = lane & 15;
    const int hi   = lane >> 4;
    const int m0 = blockIdx.y * 128;
    const int n0 = blockIdx.x * 64;

    v8f acc[4];
    for (int ct = 0; ct < 4; ++ct) acc[ct] = zero8();

#ifdef HAVE_TDM
    // One DMA for the whole B panel; wave 0 issues, everyone syncs.
    if (w == 0) {
        tdm_load_2d_f16(BT + (long)n0 * K, (unsigned)(uintptr_t)Bp,
                        /*tile_d0=*/K, /*tile_d1=*/64, /*stride=*/K);
        __builtin_amdgcn_s_wait_tensorcnt(0);
    }
    __syncthreads();
#endif

    for (int k0 = 0; k0 < K; k0 += 32) {
        // ---- stage A tile (128 x 32 halves = 512 x 8-half chunks) ----
        for (int it = 0; it < 2; ++it) {
            int ci = tid + it * 256;
            int r = ci >> 2, ch = ci & 3;
            if constexpr (AF32) {
                const float* s = (const float*)Aptr + (long)(m0 + r) * K + k0 + ch * 8;
                float4 s0 = *(const float4*)s;
                float4 s1 = *(const float4*)(s + 4);
                v8h dv;
                dv[0] = (_Float16)s0.x; dv[1] = (_Float16)s0.y;
                dv[2] = (_Float16)s0.z; dv[3] = (_Float16)s0.w;
                dv[4] = (_Float16)s1.x; dv[5] = (_Float16)s1.y;
                dv[6] = (_Float16)s1.z; dv[7] = (_Float16)s1.w;
                *(v8h*)&Al[r * 32 + ch * 8] = dv;
            } else {
                const _Float16* s = (const _Float16*)Aptr + (long)(m0 + r) * K + k0 + ch * 8;
#ifdef HAVE_ASYNC
                async_b128(s, &Al[r * 32 + ch * 8]);   // global -> LDS, no VGPRs
#else
                *(v8h*)&Al[r * 32 + ch * 8] = *(const v8h*)s;
#endif
            }
        }
#ifndef HAVE_TDM
        // ---- stage BT tile (64 x 32 halves) ----
        {
            int r = tid >> 2, ch = tid & 3;
            const _Float16* s = BT + (long)(n0 + r) * K + k0 + ch * 8;
#ifdef HAVE_ASYNC
            async_b128(s, &Bp[r * 32 + ch * 8]);
#else
            *(v8h*)&Bp[r * 32 + ch * 8] = *(const v8h*)s;
#endif
        }
#endif
#ifdef HAVE_ASYNC
        if constexpr (!AF32) __builtin_amdgcn_s_wait_asynccnt(0);
#ifndef HAVE_TDM
        else __builtin_amdgcn_s_wait_asynccnt(0);
#endif
#endif
        __syncthreads();

        // ---- WMMA (A frag and B frag both contiguous k = 16*hi + s) ----
        Frag a;
        const int arow = (w * 16 + lm) * 32;
        a.h[0] = *(v8h*)&Al[arow + hi * 16];
        a.h[1] = *(v8h*)&Al[arow + hi * 16 + 8];
        for (int ct = 0; ct < 4; ++ct) {
            Frag b;
#ifdef HAVE_TDM
            const int brow = (ct * 16 + lm) * 1024 + k0;
#else
            const int brow = (ct * 16 + lm) * 32;
#endif
            b.h[0] = *(v8h*)&Bp[brow + hi * 16];
            b.h[1] = *(v8h*)&Bp[brow + hi * 16 + 8];
            acc[ct] = wmma16(a.v, b.v, acc[ct]);
        }
        __syncthreads();
    }

    // ---- store (C layout: lane -> N = lane&15, M = g + 8*(lane>>4)) ----
    for (int ct = 0; ct < 4; ++ct) {
        for (int g = 0; g < 8; ++g) {
            const int m = m0 + w * 16 + g + 8 * hi;
            const int n = n0 + ct * 16 + lm;
            const float val = acc[ct][g];
            if constexpr (MODE == 0) {
                Ch[((long)(n >> 8) << 22) + (long)m * 256 + (n & 255)] = (_Float16)val;
            } else if constexpr (MODE == 1) {
                const long idx = ((long)((n >> 8) * 16 + (m >> 10)) * 256 + (n & 255)) * 1024
                                 + (m & 1023);
                Ch[idx] = (_Float16)val;
            } else {
                const long idx = (long)m * 1024 + n;
                Cf[idx] = val + bias[n] + resid[idx];
            }
        }
    }
}
#define GEMM_DYN_LDS (64 * 1024 * 2)   // B panel (TDM path); superset of fallback

// ---------------------------------------------------------------------------
// Fused attention: per block = (32 query rows) x (one head,batch pair).
// LDS: Qs 16KB | S 128KB (f32 scores) | P 64KB (f16 probs) = 208KB.
// Scores never touch HBM; attns f32 goes straight to d_out.
// ---------------------------------------------------------------------------
#define ATTN_LDS_BYTES (32*256*2 + 32*1024*4 + 32*1024*2)

__global__ __launch_bounds__(256) void k_attn(
    const _Float16* __restrict__ q_s, const _Float16* __restrict__ k_s,
    const _Float16* __restrict__ v_t,
    float* __restrict__ attns_out, _Float16* __restrict__ out_cat) {

    extern __shared__ char smem[];
    _Float16* Qs = (_Float16*)smem;                                  // 32 x 256
    float*    S  = (float*)(smem + 32 * 256 * 2);                    // 32 x 1024
    _Float16* P  = (_Float16*)(smem + 32 * 256 * 2 + 32 * 1024 * 4); // 32 x 1024

    const int hb = blockIdx.y;            // h*16 + b
    const int h  = hb >> 4;
    const int b  = hb & 15;
    const int q0 = blockIdx.x * 32;
    const int tid  = threadIdx.x;
    const int w    = tid >> 5;
    const int lane = tid & 31;
    const int lm   = lane & 15;
    const int hi   = lane >> 4;
    const long qkbase = (long)hb * SEQ_L * D_K;

    // ---- stage Q tile (32 x 256 halves): async global->LDS when available ----
    for (int it = 0; it < 4; ++it) {
        int ci = tid + it * 256;
        int r = ci >> 5, ch = ci & 31;
        const _Float16* s = &q_s[qkbase + (long)(q0 + r) * 256 + ch * 8];
#ifdef HAVE_ASYNC
        async_b128(s, &Qs[r * 256 + ch * 8]);
#else
        *(v8h*)&Qs[r * 256 + ch * 8] = *(const v8h*)s;
#endif
    }
#ifdef HAVE_ASYNC
    __builtin_amdgcn_s_wait_asynccnt(0);
#endif
    __syncthreads();

    // ---- Phase A: S = Q K^T / sqrt(d_model) ----
    {
        const int rt  = w >> 2;
        const int ctb = (w & 3) * 16;
        for (int j = 0; j < 16; ++j) {
            const int ct = ctb + j;
            v8f c = zero8();
            const _Float16* krow = &k_s[qkbase + (long)(ct * 16 + lm) * 256];
            for (int ks = 0; ks < 8; ++ks) {
                Frag a;
                const int arow = (rt * 16 + lm) * 256 + ks * 32;
                a.h[0] = *(v8h*)&Qs[arow + hi * 16];
                a.h[1] = *(v8h*)&Qs[arow + hi * 16 + 8];
                Frag bf;
                const _Float16* kp = krow + ks * 32 + hi * 16;
                __builtin_prefetch(kp + 64, 0, 0);
                bf.h[0] = *(const v8h*)kp;
                bf.h[1] = *(const v8h*)(kp + 8);
                c = wmma16(a.v, bf.v, c);
            }
            for (int g = 0; g < 8; ++g)
                S[(rt * 16 + g + 8 * hi) * 1024 + ct * 16 + lm] = c[g] * 0.03125f;
        }
    }
    __syncthreads();

    // ---- Phase B: stable softmax; f32 attns -> d_out, f16 probs -> LDS ----
    for (int rr = 0; rr < 4; ++rr) {
        const int r = w * 4 + rr;
        float x[32];
        float mx = -3.0e38f;
        for (int i = 0; i < 32; ++i) {
            x[i] = S[r * 1024 + lane + 32 * i];
            mx = fmaxf(mx, x[i]);
        }
        for (int o = 16; o > 0; o >>= 1) mx = fmaxf(mx, __shfl_xor(mx, o, 32));
        float sum = 0.f;
        for (int i = 0; i < 32; ++i) { x[i] = __expf(x[i] - mx); sum += x[i]; }
        for (int o = 16; o > 0; o >>= 1) sum += __shfl_xor(sum, o, 32);
        const float inv = 1.0f / sum;
        const long arow = ((long)hb * 1024 + q0 + r) * 1024;
        for (int i = 0; i < 32; ++i) {
            const float p = x[i] * inv;
            attns_out[arow + lane + 32 * i] = p;
            P[r * 1024 + lane + 32 * i] = (_Float16)p;
        }
    }
    __syncthreads();

    // ---- Phase C: O = P @ V from pre-transposed v_t ----
    {
        const int rt  = w & 1;
        const int ctb = (w >> 1) * 4;
        const long vbase = (long)hb * 256 * 1024;
        for (int j = 0; j < 4; ++j) {
            const int ct = ctb + j;
            v8f c = zero8();
            const _Float16* vrow = &v_t[vbase + (long)(ct * 16 + lm) * 1024];
            for (int ks = 0; ks < 32; ++ks) {
                Frag a;
                const int arow = (rt * 16 + lm) * 1024 + ks * 32;
                a.h[0] = *(v8h*)&P[arow + hi * 16];
                a.h[1] = *(v8h*)&P[arow + hi * 16 + 8];
                Frag bf;
                const _Float16* vp = vrow + ks * 32 + hi * 16;
                __builtin_prefetch(vp + 64, 0, 0);
                bf.h[0] = *(const v8h*)vp;
                bf.h[1] = *(const v8h*)(vp + 8);
                c = wmma16(a.v, bf.v, c);
            }
            for (int g = 0; g < 8; ++g) {
                const int m = q0 + rt * 16 + g + 8 * hi;
                const int n = ct * 16 + lm;
                out_cat[((long)b * 1024 + m) * 1024 + h * 256 + n] = (_Float16)c[g];
            }
        }
    }
}

// ---------------------------------------------------------------------------
// LayerNorm (reference: ddof=1 std, (sigma+eps) denominator)
// ---------------------------------------------------------------------------
__global__ __launch_bounds__(256) void k_ln(
    const float* __restrict__ Z, const float* __restrict__ ga,
    const float* __restrict__ gb, float* __restrict__ out) {
    __shared__ float red[8];
    const int row = blockIdx.x, tid = threadIdx.x;
    const int w = tid >> 5, lane = tid & 31;
    const float* zr = Z + (long)row * 1024;

    float v[4]; float s = 0.f;
    for (int i = 0; i < 4; ++i) { v[i] = zr[tid + 256 * i]; s += v[i]; }
    for (int o = 16; o > 0; o >>= 1) s += __shfl_xor(s, o, 32);
    if (!lane) red[w] = s;
    __syncthreads();
    float tot = 0.f;
    for (int i = 0; i < 8; ++i) tot += red[i];
    const float mu = tot * (1.0f / 1024.0f);
    __syncthreads();

    float s2 = 0.f;
    for (int i = 0; i < 4; ++i) { float d = v[i] - mu; s2 += d * d; }
    for (int o = 16; o > 0; o >>= 1) s2 += __shfl_xor(s2, o, 32);
    if (!lane) red[w] = s2;
    __syncthreads();
    float t2 = 0.f;
    for (int i = 0; i < 8; ++i) t2 += red[i];
    const float sigma = sqrtf(t2 * (1.0f / 1023.0f));
    const float inv = 1.0f / (sigma + 1e-3f);

    for (int i = 0; i < 4; ++i) {
        const int c = tid + 256 * i;
        out[(long)row * 1024 + c] = (v[i] - mu) * inv * ga[c] + gb[c];
    }
}

// ---------------------------------------------------------------------------
// Host launcher
// ---------------------------------------------------------------------------
extern "C" void kernel_launch(void* const* d_in, const int* in_sizes, int n_in,
                              void* d_out, int out_size, void* d_ws, size_t ws_size,
                              hipStream_t stream) {
    const float* q      = (const float*)d_in[0];
    const float* k      = (const float*)d_in[1];
    const float* v      = (const float*)d_in[2];
    const float* w_qs   = (const float*)d_in[3];
    const float* w_ks   = (const float*)d_in[4];
    const float* w_vs   = (const float*)d_in[5];
    const float* proj_w = (const float*)d_in[6];
    const float* proj_b = (const float*)d_in[7];
    const float* ln_a   = (const float*)d_in[8];
    const float* ln_b   = (const float*)d_in[9];

    char* ws = (char*)d_ws;
    size_t off = 0;
    auto alloc = [&](size_t bytes) -> char* {
        char* p = ws + off;
        off += (bytes + 255) & ~(size_t)255;
        return p;
    };
    _Float16* wqT = (_Float16*)alloc((size_t)N_HEAD * D_K * D_MODEL * 2);
    _Float16* wkT = (_Float16*)alloc((size_t)N_HEAD * D_K * D_MODEL * 2);
    _Float16* wvT = (_Float16*)alloc((size_t)N_HEAD * D_K * D_MODEL * 2);
    _Float16* wpT = (_Float16*)alloc((size_t)D_MODEL * D_MODEL * 2);
    _Float16* q_s = (_Float16*)alloc((size_t)N_HEAD * MROWS * D_K * 2);
    _Float16* k_s = (_Float16*)alloc((size_t)N_HEAD * MROWS * D_K * 2);
    _Float16* v_t = (_Float16*)alloc((size_t)N_HEAD * MROWS * D_K * 2);
    _Float16* cat = (_Float16*)alloc((size_t)MROWS * D_MODEL * 2);
    float*    Z   = (float*)   alloc((size_t)MROWS * D_MODEL * 4);

    float* out_main  = (float*)d_out;
    float* out_attns = out_main + OUT1_ELEMS;

    // 1) weight transpose + f32->f16
    k_transpose_cvt<<<4096, 256, 0, stream>>>(w_qs,   wqT, N_HEAD, D_MODEL, D_K);
    k_transpose_cvt<<<4096, 256, 0, stream>>>(w_ks,   wkT, N_HEAD, D_MODEL, D_K);
    k_transpose_cvt<<<4096, 256, 0, stream>>>(w_vs,   wvT, N_HEAD, D_MODEL, D_K);
    k_transpose_cvt<<<4096, 256, 0, stream>>>(proj_w, wpT, 1, D_MODEL, D_MODEL);

    // 2) QKV projections (M=16384, N=1024, K=1024); B panel via TDM when present
    dim3 gp(D_MODEL / 64, MROWS / 128);
    k_gemm<0, true><<<gp, 256, GEMM_DYN_LDS, stream>>>(q, wqT, nullptr, q_s, nullptr, nullptr);
    k_gemm<0, true><<<gp, 256, GEMM_DYN_LDS, stream>>>(k, wkT, nullptr, k_s, nullptr, nullptr);
    k_gemm<1, true><<<gp, 256, GEMM_DYN_LDS, stream>>>(v, wvT, nullptr, v_t, nullptr, nullptr);

    // 3) fused scores/softmax/PV (scores resident in 208KB of WGP LDS)
    k_attn<<<dim3(SEQ_L / 32, N_HEAD * BATCH), 256, ATTN_LDS_BYTES, stream>>>(
        q_s, k_s, v_t, out_attns, cat);

    // 4) output projection + bias + residual, then LayerNorm
    k_gemm<2, false><<<gp, 256, GEMM_DYN_LDS, stream>>>(cat, wpT, Z, nullptr, proj_b, q);
    k_ln<<<MROWS, 256, 0, stream>>>(Z, ln_a, ln_b, out_main);
}
